// DecoderRNN_549755814585
// MI455X (gfx1250) — matrix operand order repaired and checked by reading, compiled, not run on hardware
//
#include <hip/hip_runtime.h>
#include <stdint.h>

// ---------------------------------------------------------------------------
// Types
// ---------------------------------------------------------------------------
typedef __bf16 bf16_t;
typedef __attribute__((ext_vector_type(16))) __bf16 v16bf;
typedef __attribute__((ext_vector_type(8)))  __bf16 v8bf;
typedef __attribute__((ext_vector_type(8)))  float  v8f;
typedef __attribute__((ext_vector_type(4)))  unsigned int u32x4;
typedef __attribute__((ext_vector_type(8)))  int i32x8;
typedef __attribute__((ext_vector_type(4)))  int i32x4;

#define HID    512
#define BATCH  16
#define TSTEPS 256
#define VOCAB  32000
#define GDIM   2048          // 4*H gate width

// ---------------------------------------------------------------------------
// Helpers
// ---------------------------------------------------------------------------
__device__ __forceinline__ bf16_t f2bf(float f) {
    unsigned u = __float_as_uint(f);
    unsigned r = (u + 0x7FFFu + ((u >> 16) & 1u)) >> 16;
    unsigned short us = (unsigned short)r;
    return __builtin_bit_cast(bf16_t, us);
}

__device__ __forceinline__ float sigmoidf_(float x) {
    return 1.0f / (1.0f + __expf(-x));
}

// A-fragment (16x32 bf16, M rows = lane&15) from a row-major [m][k] buffer.
// lanes 0-15 : V0-3 = K[kb..kb+7],    V4-7 = K[kb+16..kb+23]
// lanes 16-31: V0-3 = K[kb+8..kb+15], V4-7 = K[kb+24..kb+31]
__device__ __forceinline__ v16bf load_a_frag(const bf16_t* base, int stride,
                                             int kb, int lane) {
    int m  = lane & 15;
    int kh = (lane >> 4) ? 8 : 0;
    const bf16_t* p = base + (size_t)m * stride + kb + kh;
    v8bf lo = *(const v8bf*)(p);
    v8bf hi = *(const v8bf*)(p + 16);
    v16bf a;
#pragma unroll
    for (int i = 0; i < 8; ++i) { a[i] = lo[i]; a[i + 8] = hi[i]; }
    return a;
}

// B-fragment (32x16 bf16): lane holds output column n = nrow0+(lane&15),
// half 0 -> K[kb..kb+7],K[kb+16..kb+23]; half 1 -> K[kb+8..kb+15],K[kb+24..kb+31].
// Works directly on row-major weights W[n][k] (rows = gate/vocab index).
__device__ __forceinline__ v16bf load_b_frag(const bf16_t* base, int stride,
                                             int nrow0, int kb, int lane) {
    int n  = lane & 15;
    int kh = (lane >> 4) ? 8 : 0;
    const bf16_t* p = base + (size_t)(nrow0 + n) * stride + kb + kh;
    v8bf lo = *(const v8bf*)(p);
    v8bf hi = *(const v8bf*)(p + 16);
    v16bf b;
#pragma unroll
    for (int i = 0; i < 8; ++i) { b[i] = lo[i]; b[i + 8] = hi[i]; }
    return b;
}

__device__ __forceinline__ v8f wmma_bf16(v16bf a, v16bf b, v8f c) {
    return __builtin_amdgcn_wmma_f32_16x16x32_bf16(false, a, false, b,
                                                   (short)0, c, false, false);
}

// Tensor Data Mover: 2D tile (tile_k contiguous elems x tile_n rows) of a
// bf16 row-major tensor (stride_k elems/row) -> contiguous LDS block.
// 6-arg builtin form: (g0, g1, g2, g3, g_extra, cpol).
__device__ __forceinline__ void tdm_load_2d(unsigned lds_off, const bf16_t* gsrc,
                                            unsigned tile_k, unsigned tile_n,
                                            unsigned tensor_k, unsigned tensor_n,
                                            unsigned stride_k) {
    unsigned long long ga = (unsigned long long)gsrc;
    u32x4 g0;
    g0.x = 1u;                                                // count=1 (valid)
    g0.y = lds_off;                                           // lds_addr
    g0.z = (unsigned)(ga & 0xFFFFFFFFull);                    // global_addr lo
    g0.w = (unsigned)((ga >> 32) & 0x1FFFFFFull) | (2u << 30);// addr hi | type=2
    i32x8 g1;
    g1[0] = (int)(1u << 16);                                  // data_size=1 (2B)
    g1[1] = (int)((tensor_k & 0xFFFFu) << 16);                // tensor_dim0 lo16
    g1[2] = (int)(((tensor_k >> 16) & 0xFFFFu) |
                  ((tensor_n & 0xFFFFu) << 16));              // dim0 hi | dim1 lo
    g1[3] = (int)(((tensor_n >> 16) & 0xFFFFu) |
                  ((tile_k & 0xFFFFu) << 16));                // dim1 hi | tile_dim0
    g1[4] = (int)(tile_n & 0xFFFFu);                          // tile_dim1 (dim2=0)
    g1[5] = (int)stride_k;                                    // dim0_stride lo32
    g1[6] = 0;
    g1[7] = 0;
    i32x4 z4 = {0, 0, 0, 0};
    i32x8 z8 = {0, 0, 0, 0, 0, 0, 0, 0};
    __builtin_amdgcn_tensor_load_to_lds(g0, g1, z4, z4, z8, 0);
}

// ---------------------------------------------------------------------------
// Prep kernels
// ---------------------------------------------------------------------------
__global__ void cvt_bf16_kernel(const float* __restrict__ s,
                                unsigned short* __restrict__ d, int n) {
    for (int i = blockIdx.x * blockDim.x + threadIdx.x; i < n;
         i += gridDim.x * blockDim.x) {
        unsigned u = __float_as_uint(s[i]);
        d[i] = (unsigned short)((u + 0x7FFFu + ((u >> 16) & 1u)) >> 16);
    }
}

__global__ void vec_add_kernel(const float* __restrict__ a,
                               const float* __restrict__ b,
                               float* __restrict__ c, int n) {
    int i = blockIdx.x * blockDim.x + threadIdx.x;
    if (i < n) c[i] = a[i] + b[i];
}

// ---------------------------------------------------------------------------
// Persistent recurrence: one workgroup (16 waves) per LSTM cell chain.
// blockIdx.x: 0=l0_f 1=l0_b 2=l1_f 3=l1_b.  c-state lives in VGPRs, h in LDS.
// Layer0 publishes y0(t) to global + release flag; layer1 acquire-spins.
// ---------------------------------------------------------------------------
__global__ __launch_bounds__(512) void lstm_recurrence_kernel(
    const float*  __restrict__ target,   // (16,256,1)
    const float*  __restrict__ enc_h,    // (4,16,512)
    const float*  __restrict__ enc_c,    // (4,16,512)
    const float*  __restrict__ wih0_f,   // (2048)
    const float*  __restrict__ wih0_b,   // (2048)
    const float*  __restrict__ bias,     // (4,2048) bih+bhh
    const bf16_t* __restrict__ whh,      // (4,2048,512)  bf16 row-major
    const bf16_t* __restrict__ wih1,     // (2,2048,1024) bf16 row-major
    bf16_t* __restrict__ y0,             // (256,16,1024)
    bf16_t* __restrict__ y1,             // (256,16,1024)
    int* __restrict__ flags,             // (256)
    float* __restrict__ out_hc)          // d_out + B*T*V : h(4,16,512) c(...)
{
    __shared__ __align__(16) bf16_t s_h[BATCH * HID];    // 16 KB, this cell's h

    const int tid   = threadIdx.x;
    const int lane  = tid & 31;
    const int wave  = tid >> 5;          // 0..15
    const int cell  = blockIdx.x;        // 0..3
    const int layer = cell >> 1;
    const int mrow  = (lane >> 4) * 8;   // accumulator M base for this lane half
    const int ncol  = lane & 15;

    const bf16_t* Whh_c  = whh + (size_t)cell * GDIM * HID;
    const float*  bias_c = bias + cell * GDIM;
    const float*  wih0   = (cell == 0) ? wih0_f : wih0_b;

    // ---- init h (LDS, bf16) and c (registers) from encoder state ----
    for (int i = tid; i < BATCH * HID; i += 512)
        s_h[i] = f2bf(enc_h[(size_t)cell * BATCH * HID + i]);

    v8f cst[2];
#pragma unroll
    for (int q = 0; q < 2; ++q) {
        int jt = wave + q * 16;
#pragma unroll
        for (int r = 0; r < 8; ++r) {
            int m = mrow + r;
            int j = jt * 16 + ncol;
            cst[q][r] = enc_c[(size_t)cell * BATCH * HID + m * HID + j];
        }
    }
    __syncthreads();

    for (int t = 0; t < TSTEPS; ++t) {
        // ---- accumulator init: bias (+ x_t * Wih for layer 0) ----
        v8f acc[2][4];
#pragma unroll
        for (int q = 0; q < 2; ++q) {
            int jt = wave + q * 16;
#pragma unroll
            for (int g = 0; g < 4; ++g) {
                int n = (g * 32 + jt) * 16 + ncol;
                float bb = bias_c[n];
                if (layer == 0) {
                    float wv = wih0[n];
#pragma unroll
                    for (int r = 0; r < 8; ++r)
                        acc[q][g][r] = bb + wv * target[(mrow + r) * TSTEPS + t];
                } else {
#pragma unroll
                    for (int r = 0; r < 8; ++r) acc[q][g][r] = bb;
                }
            }
        }

        // ---- layer1: wait for both layer0 producers, add y0 @ Wih1^T ----
        if (layer == 1) {
            if (tid == 0) {
                while (__hip_atomic_load(&flags[t], __ATOMIC_ACQUIRE,
                                         __HIP_MEMORY_SCOPE_AGENT) < 2)
                    __builtin_amdgcn_s_sleep(1);
            }
            __syncthreads();
            const bf16_t* Wih1_c = wih1 + (size_t)(cell - 2) * GDIM * 1024;
            const bf16_t* y0t    = y0 + (size_t)t * BATCH * 1024;
            for (int kb = 0; kb < 1024; kb += 32) {
                v16bf a = load_a_frag(y0t, 1024, kb, lane);
#pragma unroll
                for (int q = 0; q < 2; ++q) {
                    int jt = wave + q * 16;
#pragma unroll
                    for (int g = 0; g < 4; ++g) {
                        v16bf b = load_b_frag(Wih1_c, 1024, (g * 32 + jt) * 16,
                                              kb, lane);
                        acc[q][g] = wmma_bf16(a, b, acc[q][g]);
                    }
                }
            }
        }

        // ---- recurrent term: h(t-1) @ Whh^T, A from LDS ----
        for (int kb = 0; kb < HID; kb += 32) {
            v16bf a = load_a_frag((const bf16_t*)s_h, HID, kb, lane);
#pragma unroll
            for (int q = 0; q < 2; ++q) {
                int jt = wave + q * 16;
#pragma unroll
                for (int g = 0; g < 4; ++g) {
                    v16bf b = load_b_frag(Whh_c, HID, (g * 32 + jt) * 16,
                                          kb, lane);
                    acc[q][g] = wmma_bf16(a, b, acc[q][g]);
                }
            }
        }
        __syncthreads();   // everyone done reading s_h(t-1)

        // ---- activations; update c in regs, h in LDS; publish y ----
#pragma unroll
        for (int q = 0; q < 2; ++q) {
            int jt = wave + q * 16;
            int j  = jt * 16 + ncol;
#pragma unroll
            for (int r = 0; r < 8; ++r) {
                int m = mrow + r;
                float gi = acc[q][0][r], gf = acc[q][1][r];
                float gg = acc[q][2][r], go = acc[q][3][r];
                float cn = sigmoidf_(gf) * cst[q][r] +
                           sigmoidf_(gi) * tanhf(gg);
                float hn = sigmoidf_(go) * tanhf(cn);
                cst[q][r] = cn;
                bf16_t hb = f2bf(hn);
                s_h[m * HID + j] = hb;
                if (layer == 0)
                    y0[(size_t)t * BATCH * 1024 + m * 1024 + cell * HID + j] = hb;
                else
                    y1[(size_t)t * BATCH * 1024 + m * 1024 + (cell - 2) * HID + j] = hb;
                if (t == TSTEPS - 1) {
                    out_hc[(size_t)cell * BATCH * HID + m * HID + j] = hn;
                    out_hc[4ull * BATCH * HID + (size_t)cell * BATCH * HID +
                           m * HID + j] = cn;
                }
            }
        }

        if (layer == 0) {
            __threadfence();
            __syncthreads();
            if (tid == 0)
                __hip_atomic_fetch_add(&flags[t], 1, __ATOMIC_RELEASE,
                                       __HIP_MEMORY_SCOPE_AGENT);
        } else {
            __syncthreads();
        }
    }
}

// ---------------------------------------------------------------------------
// Deferred classifier GEMM: (4096 x 1024) x (1024 x 32000), bf16 WMMA.
// Block tile 64M x 256N, 8 waves (2x4), wave tile 32M x 64N.
// Wcls tiles (256 rows x 32 K, 16 KB) double-buffered into LDS via TDM.
// ---------------------------------------------------------------------------
__global__ __launch_bounds__(256) void classifier_gemm_kernel(
    const bf16_t* __restrict__ y1,    // (4096,1024) rows = t*16+b
    const bf16_t* __restrict__ wcls,  // (32000,1024) bf16
    const float*  __restrict__ bcls,  // (32000)
    float* __restrict__ out)          // (16,256,32000)
{
    __shared__ __align__(128) bf16_t s_b[2][256 * 32];   // 2 x 16 KB

    const int tid  = threadIdx.x;
    const int lane = tid & 31;
    const int wave = tid >> 5;
    const int n0   = blockIdx.x * 256;
    const int m0   = blockIdx.y * 64;
    const int wm   = (wave & 1) * 32;
    const int wn   = (wave >> 1) * 64;
    const bool issuer = (wave == 0);

    if (issuer)
        tdm_load_2d((unsigned)(unsigned long long)&s_b[0][0],
                    wcls + (size_t)n0 * 1024, 32, 256, 1024, VOCAB, 1024);

    v8f acc[2][4];
#pragma unroll
    for (int mt = 0; mt < 2; ++mt)
#pragma unroll
        for (int nt = 0; nt < 4; ++nt)
#pragma unroll
            for (int r = 0; r < 8; ++r) acc[mt][nt][r] = 0.0f;

    const bf16_t* a_base0 = y1 + (size_t)(m0 + wm) * 1024;
    const bf16_t* a_base1 = a_base0 + 16 * 1024;

    int buf = 0;
    for (int kb = 0; kb < 1024; kb += 32) {
        if (issuer) {
            if (kb + 32 < 1024) {
                tdm_load_2d((unsigned)(unsigned long long)&s_b[buf ^ 1][0],
                            wcls + (size_t)n0 * 1024 + (kb + 32),
                            32, 256, 1024, VOCAB, 1024);
                __builtin_amdgcn_s_wait_tensorcnt(1);
            } else {
                __builtin_amdgcn_s_wait_tensorcnt(0);
            }
        }
        __syncthreads();

        v16bf a0 = load_a_frag(a_base0, 1024, kb, lane);
        v16bf a1 = load_a_frag(a_base1, 1024, kb, lane);
        const bf16_t* sb = &s_b[buf][0];
#pragma unroll
        for (int nt = 0; nt < 4; ++nt) {
            v16bf b = load_b_frag(sb, 32, wn + nt * 16, 0, lane);
            acc[0][nt] = wmma_bf16(a0, b, acc[0][nt]);
            acc[1][nt] = wmma_bf16(a1, b, acc[1][nt]);
        }
        __syncthreads();
        buf ^= 1;
    }

    // epilogue: + bias, scatter to out[b][t][v] (row = t*16+b)
#pragma unroll
    for (int mt = 0; mt < 2; ++mt) {
#pragma unroll
        for (int nt = 0; nt < 4; ++nt) {
            int n = n0 + wn + nt * 16 + (lane & 15);
            float bv = bcls[n];
#pragma unroll
            for (int r = 0; r < 8; ++r) {
                int m = m0 + wm + mt * 16 + r + 8 * (lane >> 4);
                int tt = m >> 4, bb = m & 15;
                out[(size_t)bb * TSTEPS * VOCAB + (size_t)tt * VOCAB + n] =
                    acc[mt][nt][r] + bv;
            }
        }
    }
}

// ---------------------------------------------------------------------------
// Host
// ---------------------------------------------------------------------------
extern "C" void kernel_launch(void* const* d_in, const int* in_sizes, int n_in,
                              void* d_out, int out_size, void* d_ws, size_t ws_size,
                              hipStream_t stream) {
    const float* enc_h    = (const float*)d_in[0];
    const float* enc_c    = (const float*)d_in[1];
    const float* target   = (const float*)d_in[2];
    const float* Wih_l0_f = (const float*)d_in[3];
    const float* Whh_l0_f = (const float*)d_in[4];
    const float* bih_l0_f = (const float*)d_in[5];
    const float* bhh_l0_f = (const float*)d_in[6];
    const float* Wih_l0_b = (const float*)d_in[7];
    const float* Whh_l0_b = (const float*)d_in[8];
    const float* bih_l0_b = (const float*)d_in[9];
    const float* bhh_l0_b = (const float*)d_in[10];
    const float* Wih_l1_f = (const float*)d_in[11];
    const float* Whh_l1_f = (const float*)d_in[12];
    const float* bih_l1_f = (const float*)d_in[13];
    const float* bhh_l1_f = (const float*)d_in[14];
    const float* Wih_l1_b = (const float*)d_in[15];
    const float* Whh_l1_b = (const float*)d_in[16];
    const float* bih_l1_b = (const float*)d_in[17];
    const float* bhh_l1_b = (const float*)d_in[18];
    const float* Wcls     = (const float*)d_in[19];
    const float* bcls     = (const float*)d_in[20];

    char* ws = (char*)d_ws;
    size_t off = 0;
    auto walloc = [&](size_t bytes) -> void* {
        off = (off + 255) & ~(size_t)255;
        void* p = ws + off;
        off += bytes;
        return p;
    };

    unsigned short* whh_bf  = (unsigned short*)walloc(4ull * GDIM * HID * 2);
    unsigned short* wih1_bf = (unsigned short*)walloc(2ull * GDIM * 1024 * 2);
    unsigned short* wcls_bf = (unsigned short*)walloc((size_t)VOCAB * 1024 * 2);
    float*          bias    = (float*)walloc(4ull * GDIM * 4);
    unsigned short* y0      = (unsigned short*)walloc((size_t)TSTEPS * BATCH * 1024 * 2);
    unsigned short* y1      = (unsigned short*)walloc((size_t)TSTEPS * BATCH * 1024 * 2);
    int*            flags   = (int*)walloc(TSTEPS * 4);

    (void)hipMemsetAsync(flags, 0, TSTEPS * 4, stream);

    auto cvt = [&](const float* s, unsigned short* d, size_t n) {
        int blocks = (int)((n + 255) / 256);
        if (blocks > 4096) blocks = 4096;
        cvt_bf16_kernel<<<blocks, 256, 0, stream>>>(s, d, (int)n);
    };
    cvt(Whh_l0_f, whh_bf + 0ull * GDIM * HID, (size_t)GDIM * HID);
    cvt(Whh_l0_b, whh_bf + 1ull * GDIM * HID, (size_t)GDIM * HID);
    cvt(Whh_l1_f, whh_bf + 2ull * GDIM * HID, (size_t)GDIM * HID);
    cvt(Whh_l1_b, whh_bf + 3ull * GDIM * HID, (size_t)GDIM * HID);
    cvt(Wih_l1_f, wih1_bf + 0ull * GDIM * 1024, (size_t)GDIM * 1024);
    cvt(Wih_l1_b, wih1_bf + 1ull * GDIM * 1024, (size_t)GDIM * 1024);
    cvt(Wcls, wcls_bf, (size_t)VOCAB * 1024);

    vec_add_kernel<<<8, 256, 0, stream>>>(bih_l0_f, bhh_l0_f, bias + 0 * GDIM, GDIM);
    vec_add_kernel<<<8, 256, 0, stream>>>(bih_l0_b, bhh_l0_b, bias + 1 * GDIM, GDIM);
    vec_add_kernel<<<8, 256, 0, stream>>>(bih_l1_f, bhh_l1_f, bias + 2 * GDIM, GDIM);
    vec_add_kernel<<<8, 256, 0, stream>>>(bih_l1_b, bhh_l1_b, bias + 3 * GDIM, GDIM);

    float* out_hc = (float*)d_out + (size_t)BATCH * TSTEPS * VOCAB;
    lstm_recurrence_kernel<<<4, 512, 0, stream>>>(
        target, enc_h, enc_c, Wih_l0_f, Wih_l0_b, bias,
        (const bf16_t*)whh_bf, (const bf16_t*)wih1_bf,
        (bf16_t*)y0, (bf16_t*)y1, flags, out_hc);

    classifier_gemm_kernel<<<dim3(125, 64), 256, 0, stream>>>(
        (const bf16_t*)y1, (const bf16_t*)wcls_bf, bcls, (float*)d_out);
}
